// PackedNativeBitDense_70952859730267
// MI455X (gfx1250) — matrix operand order repaired and checked by reading, not compiled, not run on hardware
//
#include <hip/hip_runtime.h>

// ---------------------------------------------------------------------------
// out[4096,11008] = x[4096,4096] @ dequant(W)[4096,11008]
// Pipeline:
//   k0: convert_x   : x f32 -> xb bf16                 (33.5 MB in ws)
//   k1: dequant     : indices/codebook -> Wt[n][k] bf16 (90.2 MB in ws)
//   k2: gemm        : bf16 WMMA, 128x128 tiles, double-buffered LDS staged
//                     via GLOBAL_LOAD_ASYNC_TO_LDS_B128 (ASYNCcnt pipeline)
// Arithmetic intensity ~850 FLOP/B -> compute bound -> everything rides the
// v_wmma_f32_16x16x32_bf16 pipe; xb+Wt (124 MB) stay L2-resident (192 MB).
// ---------------------------------------------------------------------------

#define TOK   4096
#define D_IN  4096
#define F_OUT 11008
#define NBLK_PER_ROW 86      // 11008 / 128

#define TM 128               // M tile (tokens)
#define TN 128               // N tile (out features)
#define KS 64                // K staged per LDS pass
#define KP 72                // padded LDS row stride (bf16 units, 144 B)

typedef __attribute__((ext_vector_type(16))) __bf16 v16bf;
typedef __attribute__((ext_vector_type(8)))  float  v8f;

union Frag32B {              // 32-byte WMMA A/B operand
    uint4 q[2];
    v16bf v;
};

// f32 -> bf16 (RNE), pack two into a dword
__device__ __forceinline__ unsigned pack_bf16x2(float a, float b) {
    unsigned ua = __float_as_uint(a), ub = __float_as_uint(b);
    unsigned ra = (ua + 0x7FFFu + ((ua >> 16) & 1u)) >> 16;
    unsigned rb = (ub + 0x7FFFu + ((ub >> 16) & 1u)) & 0xFFFF0000u;
    return (ra & 0xFFFFu) | rb;
}

// gather from 8 register-resident codebook entries (v_cndmask tree)
__device__ __forceinline__ float sel8(const float c[8], int i) {
    float a0 = (i & 1) ? c[1] : c[0];
    float a1 = (i & 1) ? c[3] : c[2];
    float a2 = (i & 1) ? c[5] : c[4];
    float a3 = (i & 1) ? c[7] : c[6];
    float b0 = (i & 2) ? a1 : a0;
    float b1 = (i & 2) ? a3 : a2;
    return (i & 4) ? b1 : b0;
}

// CDNA5 async copy: 16 B global -> LDS, tracked with ASYNCcnt (ISA 10.x/15.18)
__device__ __forceinline__ void async_ld_b128(unsigned lds_byte, const void* g) {
    asm volatile("global_load_async_to_lds_b128 %0, %1, off"
                 :: "v"(lds_byte), "v"((unsigned long long)(uintptr_t)g)
                 : "memory");
}
#define WAIT_ASYNC(n) asm volatile("s_wait_asynccnt %0" :: "i"(n) : "memory")

// ---------------------------------------------------------------------------
// Kernel 0: x f32 -> bf16 (one shot; moves conversion off the GEMM hot loop)
// grid = TOK*D_IN/(256*8) = 8192, block = 256
// ---------------------------------------------------------------------------
__global__ __launch_bounds__(256) void convert_x_kernel(
    const float* __restrict__ x, unsigned short* __restrict__ xb)
{
    size_t i = ((size_t)blockIdx.x * 256 + threadIdx.x) * 8;
    float4 a = *(const float4*)(x + i);
    float4 b = *(const float4*)(x + i + 4);
    uint4 v;
    v.x = pack_bf16x2(a.x, a.y);
    v.y = pack_bf16x2(a.z, a.w);
    v.z = pack_bf16x2(b.x, b.y);
    v.w = pack_bf16x2(b.z, b.w);
    *(uint4*)(xb + i) = v;
}

// ---------------------------------------------------------------------------
// Kernel 1: dequantize + transpose.  grid = (86 nb, 32 kb), block = 256.
// Writes Wt[n][k] bf16 (row stride D_IN) so GEMM B-fragments are contiguous.
// ---------------------------------------------------------------------------
__global__ __launch_bounds__(256) void dequant_kernel(
    const int* __restrict__ idx, const float* __restrict__ cb,
    unsigned short* __restrict__ Wt)
{
    __shared__ unsigned short T[128 * 132];   // [klocal][nlocal], pad to 132
    const int tid = threadIdx.x;
    const int nb  = blockIdx.x;               // 0..85
    const int kb  = blockIdx.y;               // 0..31

    {   // phase 1: coalesced int4 index loads, one codebook block per row
        const int kl = tid >> 1;
        const int nc = (tid & 1) * 64;
        const int k  = kb * 128 + kl;
        const size_t blk = (size_t)k * NBLK_PER_ROW + nb;
        const float* c8 = cb + blk * 8;
        float4 c0 = *(const float4*)(c8);
        float4 c1 = *(const float4*)(c8 + 4);
        const float cbv[8] = {c0.x, c0.y, c0.z, c0.w, c1.x, c1.y, c1.z, c1.w};

        const int* ip = idx + (size_t)k * F_OUT + nb * 128 + nc;
        unsigned short* dst = T + kl * 132 + nc;
#pragma unroll
        for (int q = 0; q < 16; ++q) {
            int4 ii = *(const int4*)(ip + q * 4);
            uint2 p;
            p.x = pack_bf16x2(sel8(cbv, ii.x), sel8(cbv, ii.y));
            p.y = pack_bf16x2(sel8(cbv, ii.z), sel8(cbv, ii.w));
            *(uint2*)(dst + q * 4) = p;
        }
    }
    __syncthreads();
    {   // phase 2: transposed, coalesced b128 stores to Wt[n][k]
        const int nl = tid >> 1;
        const int kc = (tid & 1) * 64;
        unsigned short* op = Wt + (size_t)(nb * 128 + nl) * D_IN + kb * 128 + kc;
#pragma unroll
        for (int q = 0; q < 8; ++q) {
            unsigned short s[8];
#pragma unroll
            for (int j = 0; j < 8; ++j)
                s[j] = T[(kc + q * 8 + j) * 132 + nl];
            uint4 v;
            v.x = s[0] | ((unsigned)s[1] << 16);
            v.y = s[2] | ((unsigned)s[3] << 16);
            v.z = s[4] | ((unsigned)s[5] << 16);
            v.w = s[6] | ((unsigned)s[7] << 16);
            *(uint4*)(op + q * 8) = v;
        }
    }
}

// ---------------------------------------------------------------------------
// Kernel 2: bf16 WMMA GEMM, async-to-LDS double buffered.
// grid = 32*86 (mb fastest => W-tile L2 reuse), block = 256 (8 wave32).
// Each wave: 32x64 sub-tile = 2x4 v_wmma_f32_16x16x32_bf16 accumulators.
// ---------------------------------------------------------------------------
__global__ __launch_bounds__(256) void gemm_wmma_bf16(
    const unsigned short* __restrict__ xb, const unsigned short* __restrict__ Wt,
    float* __restrict__ out)
{
    __shared__ unsigned short Al[2][TM * KP];   // [buf][m][k]
    __shared__ unsigned short Bl[2][TN * KP];   // [buf][n][k]

    const int tid  = threadIdx.x;
    const int wg   = blockIdx.x;
    const int mb   = wg & 31;
    const int nb   = wg >> 5;
    const int m0   = mb * TM;
    const int n0   = nb * TN;

    const int lane = tid & 31;
    const int wid  = tid >> 5;
    const int wm   = wid & 3;                  // 4 wave-rows x 32
    const int wn   = wid >> 2;                 // 2 wave-cols x 64
    const int half = lane >> 4;
    const int r    = lane & 15;

    v8f c[2][4];
#pragma unroll
    for (int sm = 0; sm < 2; ++sm)
#pragma unroll
        for (int sn = 0; sn < 4; ++sn)
            c[sm][sn] = (v8f)(0.0f);

    // stage coords: thread -> (row, 32-wide k chunk); 4 b128 async ops each
    const int sr = tid >> 1;                   // 0..127
    const int sc = (tid & 1) * 32;             // 0 or 32
    const unsigned short* xrow = xb + (size_t)(m0 + sr) * D_IN + sc;
    const unsigned short* wrow = Wt + (size_t)(n0 + sr) * D_IN + sc;

    auto stage = [&](int ks, int buf) {
        const unsigned short* xp = xrow + ks * KS;
        const unsigned short* wp = wrow + ks * KS;
        unsigned al = (unsigned)(uintptr_t)&Al[buf][sr * KP + sc];
        unsigned bl = (unsigned)(uintptr_t)&Bl[buf][sr * KP + sc];
#pragma unroll
        for (int q = 0; q < 4; ++q) {
            async_ld_b128(al + q * 16, xp + q * 8);
            async_ld_b128(bl + q * 16, wp + q * 8);
        }
    };

    const int NS = D_IN / KS;                  // 64 stages
    stage(0, 0);                               // prime the pipe (8 async ops)

    for (int ks = 0; ks < NS; ++ks) {
        const int buf = ks & 1;
        if (ks + 1 < NS) {
            stage(ks + 1, buf ^ 1);            // prefetch next stage
            WAIT_ASYNC(8);                     // drain *previous* stage only
        } else {
            WAIT_ASYNC(0);
        }
        __syncthreads();                       // stage(buf) visible to all

#pragma unroll
        for (int kk = 0; kk < KS; kk += 32) {
            Frag32B a[2], b[4];
#pragma unroll
            for (int sm = 0; sm < 2; ++sm) {
                const unsigned short* p =
                    &Al[buf][(wm * 32 + sm * 16 + r) * KP + kk + half * 8];
                a[sm].q[0] = *(const uint4*)(p);
                a[sm].q[1] = *(const uint4*)(p + 16);
            }
#pragma unroll
            for (int sn = 0; sn < 4; ++sn) {
                const unsigned short* p =
                    &Bl[buf][(wn * 64 + sn * 16 + r) * KP + kk + half * 8];
                b[sn].q[0] = *(const uint4*)(p);
                b[sn].q[1] = *(const uint4*)(p + 16);
            }
#pragma unroll
            for (int sm = 0; sm < 2; ++sm)
#pragma unroll
                for (int sn = 0; sn < 4; ++sn)
                    c[sm][sn] = __builtin_amdgcn_wmma_f32_16x16x32_bf16(
                        false, a[sm].v, false, b[sn].v,
                        (short)0, c[sm][sn], false, false);
        }
        __syncthreads();                       // buf fully read before reuse
    }

    // C layout: VGPR j -> M = j + 8*half, N = lane&15
    const int rowbase = m0 + wm * 32;
    const int colbase = n0 + wn * 64;
#pragma unroll
    for (int sm = 0; sm < 2; ++sm)
#pragma unroll
        for (int sn = 0; sn < 4; ++sn) {
            float* op = out + (size_t)(rowbase + sm * 16 + half * 8) * F_OUT
                            + colbase + sn * 16 + r;
#pragma unroll
            for (int j = 0; j < 8; ++j)
                op[(size_t)j * F_OUT] = c[sm][sn][j];
        }
}

// ---------------------------------------------------------------------------
extern "C" void kernel_launch(void* const* d_in, const int* in_sizes, int n_in,
                              void* d_out, int out_size, void* d_ws, size_t ws_size,
                              hipStream_t stream) {
    const float* x        = (const float*)d_in[0];
    const int*   indices  = (const int*)d_in[1];
    const float* codebook = (const float*)d_in[2];
    float*       out      = (float*)d_out;

    // ws layout: Wt bf16 [11008][4096] = 90,177,536 B ; xb bf16 [4096][4096]
    unsigned short* Wt = (unsigned short*)d_ws;
    unsigned short* xb = (unsigned short*)((char*)d_ws
                         + (size_t)F_OUT * D_IN * sizeof(unsigned short));

    convert_x_kernel<<<dim3((TOK * D_IN) / (256 * 8)), 256, 0, stream>>>(x, xb);
    dequant_kernel<<<dim3(NBLK_PER_ROW, D_IN / 128), 256, 0, stream>>>(
        indices, codebook, Wt);
    gemm_wmma_bf16<<<dim3((TOK / TM) * (F_OUT / TN)), 256, 0, stream>>>(
        xb, Wt, out);
}
